// JSSPPolicyNetwork_80633716015322
// MI455X (gfx1250) — matrix operand order repaired and checked by reading, not compiled, not run on hardware
//
#include <hip/hip_runtime.h>
#include <hip/hip_bf16.h>
#include <math.h>

// ---------------- problem constants ----------------
#define B_   32
#define J_   20
#define O_   20
#define D_   256
#define H_   8
#define F_   1024
#define S_   400
#define DH_  32
#define NTOK (B_*J_*O_)   // 12800 tokens in main stream
#define NJOB (B_*J_)      // 640 tokens in final block
#define SMAX 416          // 13*32 padded key length (covers S=400)

typedef __attribute__((ext_vector_type(16))) _Float16 v16h;
typedef __attribute__((ext_vector_type(8)))  float    v8f;

union AFrag { unsigned int u[8]; v16h h; };
union CFrag { float f[8]; v8f v; };
union H16U  { _Float16 f; unsigned short u; };

__device__ __forceinline__ float wave_red(float x) {
#pragma unroll
  for (int off = 16; off; off >>= 1) x += __shfl_xor(x, off, 32);
  return x;
}

// ---------------- f32 -> f16 weight conversion ----------------
__global__ __launch_bounds__(256) void f32_to_f16_kernel(
    const float* __restrict__ in, _Float16* __restrict__ out, int n) {
  int i = blockIdx.x * 256 + threadIdx.x;
  if (i < n) out[i] = (_Float16)in[i];
}

// ---------------- embedding + sinusoidal PE ----------------
__global__ __launch_bounds__(256) void embed_kernel(
    const float* __restrict__ ops, const float* __restrict__ ew,
    const float* __restrict__ eb, float* __restrict__ x) {
  int tok = blockIdx.x;
  int d   = threadIdx.x;
  int o   = tok % O_;
  float dv  = expf(-logf(10000.f) * (float)(d & ~1) / (float)D_);
  float ang = (float)o * dv;
  float pe  = (d & 1) ? cosf(ang) : sinf(ang);
  x[(size_t)tok * D_ + d] =
      ops[tok * 2 + 0] * ew[d * 2 + 0] + ops[tok * 2 + 1] * ew[d * 2 + 1] + eb[d] + pe;
}

// ---------------- LayerNorm (f32 in -> f16 out), one wave per token ----------------
__global__ __launch_bounds__(256) void ln_kernel(
    const float* __restrict__ x, const float* __restrict__ g,
    const float* __restrict__ b, _Float16* __restrict__ out, int ntok) {
  int wv = threadIdx.x >> 5, lane = threadIdx.x & 31;
  int tok = blockIdx.x * 8 + wv;
  if (tok >= ntok) return;
  const float* row = x + (size_t)tok * D_;
  float v[8];
#pragma unroll
  for (int e = 0; e < 8; e++) v[e] = row[lane * 8 + e];
  float s = 0.f, ss = 0.f;
#pragma unroll
  for (int e = 0; e < 8; e++) { s += v[e]; ss += v[e] * v[e]; }
  s = wave_red(s); ss = wave_red(ss);
  float mean = s * (1.f / D_);
  float var  = ss * (1.f / D_) - mean * mean;
  float rstd = rsqrtf(var + 1e-5f);
#pragma unroll
  for (int e = 0; e < 8; e++) {
    int d = lane * 8 + e;
    out[(size_t)tok * D_ + d] = (_Float16)((v[e] - mean) * rstd * g[d] + b[d]);
  }
}

// ---------------- WMMA GEMM: C[M,N] = A16[M,K] @ W16[N,K]^T + bias (+gelu)(+resid) ----
// block = 8 waves; each wave computes a 32x32 tile (2 A-frags x 2 B-frags -> 4 WMMA),
// so fragment loads per WMMA = 1.0 instead of 2.0. Block covers 32(M) x 256(N).
__global__ __launch_bounds__(256) void gemm_wmma(
    const _Float16* __restrict__ A, const _Float16* __restrict__ W,
    const float* __restrict__ bias, const float* __restrict__ resid,
    float* __restrict__ outF, _Float16* __restrict__ outH,
    int K, int N, int gelu) {
  int wv = threadIdx.x >> 5, lane = threadIdx.x & 31;
  int hl = lane >> 4, lx = lane & 15;
  int m0 = blockIdx.x * 32;
  int n0 = blockIdx.y * 256 + wv * 32;
  const _Float16* Ar0 = A + (size_t)(m0 + lx) * K;       // A frag: M = lane&15
  const _Float16* Ar1 = A + (size_t)(m0 + 16 + lx) * K;
  const _Float16* Wr0 = W + (size_t)(n0 + lx) * K;       // B frag: N = lane&15
  const _Float16* Wr1 = W + (size_t)(n0 + 16 + lx) * K;
  CFrag c00, c01, c10, c11;
#pragma unroll
  for (int v = 0; v < 8; v++) { c00.f[v] = 0.f; c01.f[v] = 0.f; c10.f[v] = 0.f; c11.f[v] = 0.f; }
  for (int k0 = 0; k0 < K; k0 += 32) {
    AFrag a0, a1, b0, b1;
#pragma unroll
    for (int v = 0; v < 8; v++) {
      int kA = ((v >> 2) << 4) + (hl << 3) + ((v & 3) << 1);  // A: 16-bit 16x32 layout
      a0.u[v] = *(const unsigned int*)(Ar0 + k0 + kA);
      a1.u[v] = *(const unsigned int*)(Ar1 + k0 + kA);
      int kB = ((v >> 2) << 4) + (hl << 3) + ((v & 3) << 1);  // B: same k order as A
      b0.u[v] = *(const unsigned int*)(Wr0 + k0 + kB);
      b1.u[v] = *(const unsigned int*)(Wr1 + k0 + kB);
    }
    if (k0 + 32 < K) {
      __builtin_prefetch(Wr0 + k0 + 32, 0, 1);
      __builtin_prefetch(Wr1 + k0 + 32, 0, 1);
    }
    c00.v = __builtin_amdgcn_wmma_f32_16x16x32_f16(false, a0.h, false, b0.h, (short)0, c00.v, false, false);
    c01.v = __builtin_amdgcn_wmma_f32_16x16x32_f16(false, a0.h, false, b1.h, (short)0, c01.v, false, false);
    c10.v = __builtin_amdgcn_wmma_f32_16x16x32_f16(false, a1.h, false, b0.h, (short)0, c10.v, false, false);
    c11.v = __builtin_amdgcn_wmma_f32_16x16x32_f16(false, a1.h, false, b1.h, (short)0, c11.v, false, false);
  }
#pragma unroll
  for (int mi = 0; mi < 2; mi++) {
#pragma unroll
    for (int ni = 0; ni < 2; ni++) {
      CFrag* c = (mi == 0) ? (ni == 0 ? &c00 : &c01) : (ni == 0 ? &c10 : &c11);
      int n = n0 + ni * 16 + lx;
      float bval = bias[n];
#pragma unroll
      for (int v = 0; v < 8; v++) {
        int m = m0 + mi * 16 + v + (hl << 3);                 // C: M = vgpr + 8*half
        float val = c->f[v] + bval;
        if (gelu) val = 0.5f * val * (1.f + erff(val * 0.70710678118f));
        if (resid) val += resid[(size_t)m * N + n];
        if (outF) outF[(size_t)m * N + n] = val;
        if (outH) outH[(size_t)m * N + n] = (_Float16)val;
      }
    }
  }
}

// ---------------- attention: one wave per (seq, head, 16-row query tile) ------------
// mode 0: job mask * ALiBi slope, seq=20 ; mode 1: machine mask, seq=400 ;
// mode 2: action-mask on keys, seq=20.
// V tile is staged into LDS (overlaying the dead score buffer) with
// global_load_async_to_lds_b32 + s_wait_asynccnt.
__global__ __launch_bounds__(32) void attn_kernel(
    const _Float16* __restrict__ qkv, _Float16* __restrict__ out,
    const float* __restrict__ jm, const float* __restrict__ mm,
    const unsigned char* __restrict__ am,
    int seqlen, int qtiles, int nkb, int mode) {
  __shared__ float          sS[16][SMAX];   // scores; reused as V tile after softmax
  __shared__ unsigned short sP[16][SMAX];   // probabilities (f16 bits)
  _Float16 (*sV)[DH_] = (_Float16 (*)[DH_])(&sS[0][0]);  // [SMAX][32] f16 == same bytes

  int lane = threadIdx.x;
  int lx = lane & 15, hl = lane >> 4;
  int bid = blockIdx.x;
  int qt = bid % qtiles;
  int sh = bid / qtiles;
  int h = sh % H_;
  int s = sh / H_;
  int tok0 = s * seqlen;
  int kpad = nkb * 32;
  const float scale = 0.17677669529f;  // 1/sqrt(32)
  float slope = 1.0f / (float)(1 << (h + 1));

  // Q fragment (A layout), rows beyond seqlen -> 0
  AFrag aq;
  int qrow = qt * 16 + lx;
#pragma unroll
  for (int v = 0; v < 8; v++) {
    int kA = ((v >> 2) << 4) + (hl << 3) + ((v & 3) << 1);
    aq.u[v] = (qrow < seqlen)
                  ? *(const unsigned int*)(qkv + (size_t)(tok0 + qrow) * 768 + h * DH_ + kA)
                  : 0u;
  }

  // scores = Q K^T * scale + mask  -> LDS
  for (int kb = 0; kb < nkb; kb++) {
#pragma unroll
    for (int sub = 0; sub < 2; sub++) {
      int kt0 = kb * 32 + sub * 16;
      AFrag bk;
#pragma unroll
      for (int v = 0; v < 8; v++) {
        int kB = ((v >> 2) << 4) + (hl << 3) + ((v & 3) << 1);
        int kk = kt0 + lx;
        bk.u[v] = (kk < seqlen)
                      ? *(const unsigned int*)(qkv + (size_t)(tok0 + kk) * 768 + D_ + h * DH_ + kB)
                      : 0u;
      }
      CFrag zc;
#pragma unroll
      for (int v = 0; v < 8; v++) zc.f[v] = 0.f;
      CFrag cs;
      cs.v = __builtin_amdgcn_wmma_f32_16x16x32_f16(false, aq.h, false, bk.h,
                                                    (short)0, zc.v, false, false);
#pragma unroll
      for (int v = 0; v < 8; v++) {
        int r = v + (hl << 3);
        int q = qt * 16 + r;
        int k = kt0 + lx;
        float val;
        if (k < seqlen) {
          int qc = (q < seqlen) ? q : 0;
          float msk;
          if (mode == 0)      msk = jm[(size_t)s * (O_ * O_) + qc * O_ + k] * slope;
          else if (mode == 1) msk = mm[(size_t)s * (S_ * S_) + qc * S_ + k];
          else                msk = am[s * J_ + k] ? -3.0e38f : 0.f;
          val = cs.f[v] * scale + msk;
        } else {
          val = -3.0e38f;
        }
        sS[r][k] = val;
      }
    }
  }
  __syncthreads();

  // per-row softmax, probs -> f16 in LDS (zero-padded to kpad)
  if (lane < 16) {
    int r = lane;
    float mx = -3.0e38f;
    for (int k = 0; k < seqlen; k++) mx = fmaxf(mx, sS[r][k]);
    float sum = 0.f;
    for (int k = 0; k < seqlen; k++) { float e = expf(sS[r][k] - mx); sS[r][k] = e; sum += e; }
    float inv = 1.f / sum;
    for (int k = 0; k < kpad; k++) {
      H16U u; u.f = (_Float16)((k < seqlen) ? sS[r][k] * inv : 0.f);
      sP[r][k] = u.u;
    }
  }
  __syncthreads();  // sS is dead from here: reuse as sV

  // zero-fill padded V rows (so 0-prob x garbage never makes NaN in WMMA)
  for (int t = seqlen; t < kpad; t++) sV[t][lane] = (_Float16)0.f;
  // async-stage V tile: lane (hl,lx) copies dword lx of token row t2+hl
  for (int t2 = 0; t2 < seqlen; t2 += 2) {
    int t = t2 + hl;
    if (t < seqlen) {
      const void* g = (const void*)(qkv + (size_t)(tok0 + t) * 768 + 2 * D_ + h * DH_ + lx * 2);
      unsigned ldsa = (unsigned)(unsigned long long)(uintptr_t)(&sV[t][lx * 2]);
      asm volatile("global_load_async_to_lds_b32 %0, %1, off"
                   :: "v"(ldsa), "v"(g) : "memory");
    }
  }
  asm volatile("s_wait_asynccnt 0x0" ::: "memory");
  __syncthreads();

  // O = P V  (DH=32 -> two 16-wide N halves), P from LDS, V from LDS
  CFrag co0, co1;
#pragma unroll
  for (int v = 0; v < 8; v++) { co0.f[v] = 0.f; co1.f[v] = 0.f; }
  for (int kb = 0; kb < nkb; kb++) {
    AFrag ap;
#pragma unroll
    for (int v = 0; v < 8; v++) {
      int kA = ((v >> 2) << 4) + (hl << 3) + ((v & 3) << 1);
      ap.u[v] = *(const unsigned int*)&sP[lx][kb * 32 + kA];
    }
#pragma unroll
    for (int hn = 0; hn < 2; hn++) {
      AFrag bv;
#pragma unroll
      for (int v = 0; v < 8; v++) {
        int kk = ((v >> 2) << 4) + (hl << 3) + ((v & 3) << 1);
        int t = kb * 32 + kk;
        int col = hn * 16 + lx;
        H16U lo, hi;
        lo.f = sV[t][col];
        hi.f = sV[t + 1][col];
        bv.u[v] = (unsigned int)lo.u | ((unsigned int)hi.u << 16);
      }
      if (hn == 0)
        co0.v = __builtin_amdgcn_wmma_f32_16x16x32_f16(false, ap.h, false, bv.h,
                                                       (short)0, co0.v, false, false);
      else
        co1.v = __builtin_amdgcn_wmma_f32_16x16x32_f16(false, ap.h, false, bv.h,
                                                       (short)0, co1.v, false, false);
    }
  }
#pragma unroll
  for (int v = 0; v < 8; v++) {
    int r = v + (hl << 3);
    int q = qt * 16 + r;
    if (q < seqlen) {
      size_t base = (size_t)(tok0 + q) * D_ + h * DH_;
      out[base + lx]      = (_Float16)co0.f[v];
      out[base + 16 + lx] = (_Float16)co1.f[v];
    }
  }
}

// ---------------- gather next-op token ----------------
__global__ __launch_bounds__(256) void gather_kernel(
    const float* __restrict__ x, const int* __restrict__ idx, float* __restrict__ nxt) {
  int t = blockIdx.x, d = threadIdx.x;
  int id = idx[t];
  id = (id < 0) ? 0 : ((id > O_ - 1) ? O_ - 1 : id);
  nxt[(size_t)t * D_ + d] = x[((size_t)t * O_ + id) * D_ + d];
}

// ---------------- policy head ----------------
__global__ __launch_bounds__(256) void policy_kernel(
    const float* __restrict__ nxt, const float* __restrict__ pw,
    const float* __restrict__ pb, const unsigned char* __restrict__ am,
    float* __restrict__ out) {
  int wv = threadIdx.x >> 5, lane = threadIdx.x & 31;
  int t = blockIdx.x * 8 + wv;
  if (t >= NJOB) return;
  float sdot = 0.f;
#pragma unroll
  for (int e = 0; e < 8; e++) {
    int d = lane * 8 + e;
    sdot += nxt[(size_t)t * D_ + d] * pw[d];
  }
  sdot = wave_red(sdot);
  if (lane == 0) out[t] = am[t] ? -__builtin_inff() : (sdot + pb[0]);
}

// ---------------- launch ----------------
extern "C" void kernel_launch(void* const* d_in, const int* in_sizes, int n_in,
                              void* d_out, int out_size, void* d_ws, size_t ws_size,
                              hipStream_t stream) {
  (void)in_sizes; (void)n_in; (void)out_size; (void)ws_size;
  const float* ops  = (const float*)d_in[0];
  const float* jm   = (const float*)d_in[1];
  const float* mm   = (const float*)d_in[2];
  const int*   jn   = (const int*)d_in[3];
  const unsigned char* am = (const unsigned char*)d_in[4];
  const float* qkvw = (const float*)d_in[5];
  const float* qkvb = (const float*)d_in[6];
  const float* outw = (const float*)d_in[7];
  const float* outb = (const float*)d_in[8];
  const float* ln1g = (const float*)d_in[9];
  const float* ln1b = (const float*)d_in[10];
  const float* ln2g = (const float*)d_in[11];
  const float* ln2b = (const float*)d_in[12];
  const float* ff1w = (const float*)d_in[13];
  const float* ff1b = (const float*)d_in[14];
  const float* ff2w = (const float*)d_in[15];
  const float* ff2b = (const float*)d_in[16];
  const float* ew   = (const float*)d_in[17];
  const float* eb   = (const float*)d_in[18];
  const float* pw   = (const float*)d_in[19];
  const float* pb   = (const float*)d_in[20];

  char* ws = (char*)d_ws;
  size_t off = 0;
  auto alloc = [&](size_t bytes) -> char* {
    char* p = ws + off;
    off += (bytes + 255) & ~(size_t)255;
    return p;
  };
  float*    xf    = (float*)   alloc((size_t)NTOK * D_ * 4);
  _Float16* h16   = (_Float16*)alloc((size_t)NTOK * D_ * 2);
  _Float16* qkv16 = (_Float16*)alloc((size_t)NTOK * 768 * 2);
  _Float16* at16  = (_Float16*)alloc((size_t)NTOK * D_ * 2);
  _Float16* ff16  = (_Float16*)alloc((size_t)NTOK * F_ * 2);
  float*    nxt   = (float*)   alloc((size_t)NJOB * D_ * 4);
  _Float16* wq16  = (_Float16*)alloc((size_t)7 * 768 * D_ * 2);
  _Float16* wo16  = (_Float16*)alloc((size_t)7 * D_ * D_ * 2);
  _Float16* w116  = (_Float16*)alloc((size_t)7 * F_ * D_ * 2);
  _Float16* w216  = (_Float16*)alloc((size_t)7 * D_ * F_ * 2);

  // one-shot weight conversion (deterministic: runs every call)
  int nq = 7 * 768 * D_;
  f32_to_f16_kernel<<<(nq + 255) / 256, 256, 0, stream>>>(qkvw, wq16, nq);
  int no = 7 * D_ * D_;
  f32_to_f16_kernel<<<(no + 255) / 256, 256, 0, stream>>>(outw, wo16, no);
  int n1 = 7 * F_ * D_;
  f32_to_f16_kernel<<<(n1 + 255) / 256, 256, 0, stream>>>(ff1w, w116, n1);
  int n2 = 7 * D_ * F_;
  f32_to_f16_kernel<<<(n2 + 255) / 256, 256, 0, stream>>>(ff2w, w216, n2);

  embed_kernel<<<NTOK, 256, 0, stream>>>(ops, ew, eb, xf);

  for (int i = 0; i < 6; i++) {
    ln_kernel<<<NTOK / 8, 256, 0, stream>>>(xf, ln1g + i * D_, ln1b + i * D_, h16, NTOK);
    gemm_wmma<<<dim3(NTOK / 32, 768 / 256), 256, 0, stream>>>(
        h16, wq16 + (size_t)i * 768 * D_, qkvb + i * 768,
        nullptr, nullptr, qkv16, D_, 768, 0);
    if ((i & 1) == 0)
      attn_kernel<<<NJOB * H_ * 2, 32, 0, stream>>>(qkv16, at16, jm, mm, am, O_, 2, 1, 0);
    else
      attn_kernel<<<B_ * H_ * 25, 32, 0, stream>>>(qkv16, at16, jm, mm, am, S_, 25, 13, 1);
    gemm_wmma<<<dim3(NTOK / 32, 1), 256, 0, stream>>>(
        at16, wo16 + (size_t)i * D_ * D_, outb + i * D_,
        xf, xf, nullptr, D_, D_, 0);
    ln_kernel<<<NTOK / 8, 256, 0, stream>>>(xf, ln2g + i * D_, ln2b + i * D_, h16, NTOK);
    gemm_wmma<<<dim3(NTOK / 32, F_ / 256), 256, 0, stream>>>(
        h16, w116 + (size_t)i * F_ * D_, ff1b + i * F_,
        nullptr, nullptr, ff16, D_, F_, 1);
    gemm_wmma<<<dim3(NTOK / 32, 1), 256, 0, stream>>>(
        ff16, w216 + (size_t)i * D_ * F_, ff2b + i * D_,
        xf, xf, nullptr, F_, D_, 0);
  }

  // final block (layer 6) over gathered next-op tokens
  gather_kernel<<<NJOB, 256, 0, stream>>>(xf, jn, nxt);
  ln_kernel<<<NJOB / 8, 256, 0, stream>>>(nxt, ln1g + 6 * D_, ln1b + 6 * D_, h16, NJOB);
  gemm_wmma<<<dim3(NJOB / 32, 768 / 256), 256, 0, stream>>>(
      h16, wq16 + (size_t)6 * 768 * D_, qkvb + 6 * 768,
      nullptr, nullptr, qkv16, D_, 768, 0);
  attn_kernel<<<B_ * H_ * 2, 32, 0, stream>>>(qkv16, at16, jm, mm, am, J_, 2, 1, 2);
  gemm_wmma<<<dim3(NJOB / 32, 1), 256, 0, stream>>>(
      at16, wo16 + (size_t)6 * D_ * D_, outb + 6 * D_,
      nxt, nxt, nullptr, D_, D_, 0);
  ln_kernel<<<NJOB / 8, 256, 0, stream>>>(nxt, ln2g + 6 * D_, ln2b + 6 * D_, h16, NJOB);
  gemm_wmma<<<dim3(NJOB / 32, F_ / 256), 256, 0, stream>>>(
      h16, w116 + (size_t)6 * F_ * D_, ff1b + 6 * F_,
      nullptr, nullptr, ff16, D_, F_, 1);
  gemm_wmma<<<dim3(NJOB / 32, 1), 256, 0, stream>>>(
      ff16, w216 + (size_t)6 * D_ * F_, ff2b + 6 * D_,
      nxt, nxt, nullptr, F_, D_, 0);

  policy_kernel<<<NJOB / 8, 256, 0, stream>>>(nxt, pw, pb, am, (float*)d_out);
}
